// AttentionWordRNN_68710886802009
// MI455X (gfx1250) — matrix operand order repaired and checked by reading, not compiled, not run on hardware
//
#include <hip/hip_runtime.h>
#include <hip/hip_bf16.h>

// ---------------------------------------------------------------------------
// AttentionWordRNN for MI455X (gfx1250): bf16 WMMA GEMMs (TDM + async-to-LDS
// staged) + persistent-LDS GRU scan + fused attention epilogue.
//
// Dims: V=50000, E=256, H=256, B=128, T=200.
// Outputs (flat in d_out, fp32): sent[128*512] | state[2*128*256] | attn[128*200]
// ---------------------------------------------------------------------------

typedef __attribute__((ext_vector_type(8)))  __bf16 v8bf;
typedef __attribute__((ext_vector_type(16))) __bf16 v16bf;
typedef __attribute__((ext_vector_type(8)))  float  v8f;
typedef __attribute__((ext_vector_type(4)))  unsigned int u32x4;
typedef __attribute__((ext_vector_type(4)))  int  i32x4;
typedef __attribute__((ext_vector_type(8)))  int  i32x8;

// pointer types matching the async-to-LDS builtin prototype
typedef __attribute__((address_space(1))) i32x4 as1_i32x4;
typedef __attribute__((address_space(3))) i32x4 as3_i32x4;

#define Bm 128
#define Tm 200
#define Em 256
#define Hm 256

// ---- feature probes -------------------------------------------------------
#if __has_builtin(__builtin_amdgcn_global_load_async_to_lds_b128)
#define HAVE_ASYNC 1
#else
#define HAVE_ASYNC 0
#endif
#if __has_builtin(__builtin_amdgcn_tensor_load_to_lds)
#define HAVE_TDM 1
#else
#define HAVE_TDM 0
#endif
#if __has_builtin(__builtin_amdgcn_s_wait_asynccnt)
#define WAIT_ASYNC(n) __builtin_amdgcn_s_wait_asynccnt(n)
#elif HAVE_ASYNC
#define WAIT_ASYNC(n) asm volatile("s_wait_asynccnt %0" ::"i"(n) : "memory")
#else
#define WAIT_ASYNC(n)
#endif
#if __has_builtin(__builtin_amdgcn_s_wait_tensorcnt)
#define WAIT_TENSOR(n) __builtin_amdgcn_s_wait_tensorcnt(n)
#else
#define WAIT_TENSOR(n)
#endif

// async copy counts per wave per chunk: B = 2 instr; A = 1 instr unless TDM
#if HAVE_TDM
#define PEND 2
#else
#define PEND 3
#endif

// ---- workspace layout (bytes, 256-aligned) --------------------------------
#define OFF_X      ((size_t)0)                       // x bf16 [T*B, E]        13,107,200
#define OFF_WIHF   ((size_t)13107200)                // w_ih_f bf16 [768,256]     393,216
#define OFF_WHHF   ((size_t)13500416)                // w_hh_f bf16 [768,256]     393,216
#define OFF_WIHB   ((size_t)13893632)
#define OFF_WHHB   ((size_t)14286848)
#define OFF_WT     ((size_t)14680064)                // W_word^T bf16 [512,512]   524,288
#define OFF_GIF    ((size_t)15204352)                // gi_f f32 [T*B,768]     78,643,200
#define OFF_GIB    ((size_t)93847552)                // gi_b f32 [T*B,768]     78,643,200
#define OFF_OW     ((size_t)172490752)               // output_word f32 [B,T,512] 52,428,800
#define OFF_OWBF   ((size_t)224919552)               // output_word bf16          26,214,400
#define OFF_LOG    ((size_t)251133952)               // attn logits f32 [B*T]        102,400

__device__ __forceinline__ __bf16 f2bf(float f) {
  union { float f; unsigned u; } v; v.f = f;
  unsigned r = v.u + 0x7FFFu + ((v.u >> 16) & 1u);   // round-to-nearest-even
  union { unsigned short s; __bf16 b; } o; o.s = (unsigned short)(r >> 16);
  return o.b;
}

__device__ __forceinline__ v16bf make_frag(v8bf lo, v8bf hi) {
  v16bf r;
#pragma unroll
  for (int i = 0; i < 8; ++i) { r[i] = lo[i]; r[8 + i] = hi[i]; }
  return r;
}

__device__ __forceinline__ v8f wmma_bf16(v16bf a, v16bf b, v8f c) {
  return __builtin_amdgcn_wmma_f32_16x16x32_bf16(false, a, false, b, (short)0, c,
                                                 false, false);
}

// 16-byte global -> LDS copy: async engine when available, sync fallback.
__device__ __forceinline__ void cp16(const __bf16* g, __bf16* l) {
#if HAVE_ASYNC
  __builtin_amdgcn_global_load_async_to_lds_b128(
      (as1_i32x4*)(void*)(__bf16*)g, (as3_i32x4*)(void*)l, 0, 0);
#else
  *(v8bf*)l = *(const v8bf*)g;
#endif
}

#if HAVE_TDM
// 2D TDM load: tile of `rows` x `kelems` bf16 from a row-major tensor with
// row stride `stride` elements, into contiguous LDS [rows][kelems].
__device__ __forceinline__ void tdm_load_2d(const void* gptr, void* lptr,
                                            unsigned kelems, unsigned rows,
                                            unsigned stride) {
  unsigned lds = (unsigned)(unsigned long long)
      (__attribute__((address_space(3))) char*)lptr;
  unsigned long long ga = (unsigned long long)gptr;
  u32x4 g0;
  g0[0] = 1u;                                      // count=1, user mode
  g0[1] = lds;                                     // lds_addr
  g0[2] = (unsigned)(ga & 0xFFFFFFFFu);            // global_addr[31:0]
  g0[3] = (unsigned)((ga >> 32) & 0x01FFFFFFu) | (2u << 30);  // [56:32] | type=2
  i32x8 g1;
  g1[0] = 1 << 16;                                 // data_size = 2 bytes
  g1[1] = (int)((stride & 0xFFFFu) << 16);         // tensor_dim0[15:0]
  g1[2] = (int)(((stride >> 16) & 0xFFFFu) | ((rows & 0xFFFFu) << 16));
  g1[3] = (int)(((rows >> 16) & 0xFFFFu) | ((kelems & 0xFFFFu) << 16)); // tile_dim0
  g1[4] = (int)(rows & 0xFFFFu);                   // tile_dim1 (tile_dim2=0)
  g1[5] = (int)stride;                             // tensor_dim0_stride[31:0]
  g1[6] = 0;
  g1[7] = 0;
  i32x4 z4 = {};
#if __clang_major__ >= 23
  i32x8 z8 = {};
  __builtin_amdgcn_tensor_load_to_lds(g0, g1, z4, z4, z8, 0);
#else
  __builtin_amdgcn_tensor_load_to_lds(g0, g1, z4, z4, 0);
#endif
}
#endif

// ---------------------------------------------------------------------------
// 1) embedding gather + fp32->bf16, x[T*B, E] with row = t*B + b
// ---------------------------------------------------------------------------
__global__ void __launch_bounds__(256)
k_gather_x(const int* __restrict__ ids, const float* __restrict__ emb,
           __bf16* __restrict__ xbf) {
  int row = blockIdx.x;                 // 0..25599 = t*B+b
  int e   = threadIdx.x;                // 0..255
  int b = row & (Bm - 1);
  int t = row >> 7;
  int id = ids[b * Tm + t];
  xbf[(size_t)row * Em + e] = f2bf(emb[(size_t)id * Em + e]);
}

// ---------------------------------------------------------------------------
// 2) convert weights to bf16 (w_ih/w_hh kept [N,K] row-major; W_word transposed)
// ---------------------------------------------------------------------------
__global__ void __launch_bounds__(256)
k_conv_w(const float* __restrict__ wihf, const float* __restrict__ whhf,
         const float* __restrict__ wihb, const float* __restrict__ whhb,
         const float* __restrict__ Wword,
         __bf16* __restrict__ o_wihf, __bf16* __restrict__ o_whhf,
         __bf16* __restrict__ o_wihb, __bf16* __restrict__ o_whhb,
         __bf16* __restrict__ o_Wt) {
  int idx = blockIdx.x * 256 + threadIdx.x;      // 0 .. 1048575
  const int NW = 3 * Hm * Em;                    // 196608
  if (idx < NW)                 { o_wihf[idx] = f2bf(wihf[idx]); return; }
  idx -= NW;
  if (idx < NW)                 { o_whhf[idx] = f2bf(whhf[idx]); return; }
  idx -= NW;
  if (idx < NW)                 { o_wihb[idx] = f2bf(wihb[idx]); return; }
  idx -= NW;
  if (idx < NW)                 { o_whhb[idx] = f2bf(whhb[idx]); return; }
  idx -= NW;                                      // 0..262143  -> Wt[j][k]=W[k][j]
  int j = idx >> 9, k = idx & 511;
  o_Wt[idx] = f2bf(Wword[(size_t)k * 512 + j]);
}

// ---------------------------------------------------------------------------
// 3) gi = x @ w_ih^T + b_ih, block-tiled with TDM (A) + async (B) staging.
//    Macro tile 64x128, K=256 in 8 chunks of 32, double-buffered in LDS.
//    8 waves: wave w -> m-tile (w>>1), n-tiles (w&1)*4 .. +3.
// ---------------------------------------------------------------------------
__global__ void __launch_bounds__(256)
k_gemm_gi(const __bf16* __restrict__ xbf,
          const __bf16* __restrict__ w_f, const __bf16* __restrict__ w_b,
          const float* __restrict__ bih_f, const float* __restrict__ bih_b,
          float* __restrict__ gi_f, float* __restrict__ gi_b) {
  __shared__ __bf16 Abuf[2][64 * 32];    // 8 KB   [row][k]
  __shared__ __bf16 Bbuf[2][128 * 32];   // 16 KB  [row][k]

  const int MB = 400, NB = 6;
  int blk = blockIdx.x;
  int dir = blk / (MB * NB);
  int t = blk - dir * (MB * NB);
  int mb = t / NB, nb = t - mb * NB;
  const __bf16* w   = dir ? w_b   : w_f;
  const float*  bih = dir ? bih_b : bih_f;
  float*        gi  = dir ? gi_b  : gi_f;

  const __bf16* Ab = xbf + (size_t)mb * 64 * Em;
  const __bf16* Bb = w   + (size_t)nb * 128 * Em;

  int tid = threadIdx.x;
  int lane = tid & 31, wave = tid >> 5;
  int half = lane >> 4, l15 = lane & 15;

  auto issueA = [&](int kc, int buf) {
#if HAVE_TDM
    if (tid < 32)
      tdm_load_2d(Ab + kc * 32, &Abuf[buf][0], 32u, 64u, (unsigned)Em);
#else
    {  // 256 x 16B cooperative copy (64 rows x 4 segments)
      int row = tid >> 2, part = tid & 3;
      cp16(Ab + row * Em + kc * 32 + part * 8, &Abuf[buf][row * 32 + part * 8]);
    }
#endif
  };
  auto issueB = [&](int kc, int buf) {
#pragma unroll
    for (int s = 0; s < 2; ++s) {
      int seg = tid + s * 256;           // 512 x 16B (128 rows x 4 segments)
      int row = seg >> 2, part = seg & 3;
      cp16(Bb + row * Em + kc * 32 + part * 8, &Bbuf[buf][row * 32 + part * 8]);
    }
  };

  int mt = wave >> 1;                // 0..3
  int ntb = (wave & 1) * 4;          // 0 or 4
  v8f acc[4] = {{}, {}, {}, {}};

  issueA(0, 0);
  issueB(0, 0);

  for (int kc = 0; kc < 8; ++kc) {
    int cbuf = kc & 1;
    if (kc < 7) {
      issueA(kc + 1, cbuf ^ 1);
      issueB(kc + 1, cbuf ^ 1);
#if HAVE_TDM
      if (wave == 0) WAIT_TENSOR(1);
#endif
      WAIT_ASYNC(PEND);
    } else {
#if HAVE_TDM
      if (wave == 0) WAIT_TENSOR(0);
#endif
      WAIT_ASYNC(0);
    }
    __syncthreads();

    const __bf16* ap = &Abuf[cbuf][(mt * 16 + l15) * 32];
    int ka = half ? 8 : 0;
    v16bf a = make_frag(*(const v8bf*)(ap + ka), *(const v8bf*)(ap + ka + 16));
#pragma unroll
    for (int j = 0; j < 4; ++j) {
      const __bf16* bp = &Bbuf[cbuf][((ntb + j) * 16 + l15) * 32];
      int kb = half ? 8 : 0;
      v16bf b = make_frag(*(const v8bf*)(bp + kb), *(const v8bf*)(bp + kb + 16));
      acc[j] = wmma_bf16(a, b, acc[j]);
    }
    __syncthreads();
  }

  int rowbase = mb * 64 + mt * 16 + half * 8;
#pragma unroll
  for (int j = 0; j < 4; ++j) {
    int n = nb * 128 + (ntb + j) * 16 + l15;
    float bias = bih[n];
#pragma unroll
    for (int i = 0; i < 8; ++i)
      gi[(size_t)(rowbase + i) * 768 + n] = acc[j][i] + bias;
  }
}

// ---------------------------------------------------------------------------
// 4) persistent GRU scan. grid.x = 2 (direction). 1024 threads = 32 waves.
//    LDS: h fp32 (128KB) + double-buffered h bf16 (2x64KB) = 256KB dynamic.
// ---------------------------------------------------------------------------
__global__ void __launch_bounds__(1024)
k_gru_scan(const float* __restrict__ gi_f, const float* __restrict__ gi_b,
           const __bf16* __restrict__ whh_f, const __bf16* __restrict__ whh_b,
           const float* __restrict__ bhh_f, const float* __restrict__ bhh_b,
           const float* __restrict__ state0,
           float* __restrict__ ow, __bf16* __restrict__ owbf,
           float* __restrict__ state_out) {
  extern __shared__ unsigned char smem[];
  float*  hf  = (float*)smem;                                   // [128][256]
  __bf16* hb0 = (__bf16*)(smem + 131072);                       // [128][256]
  __bf16* hb1 = (__bf16*)(smem + 131072 + 65536);

  int dir = blockIdx.x;
  const float*  gi  = dir ? gi_b  : gi_f;
  const __bf16* whh = dir ? whh_b : whh_f;
  const float*  bhh = dir ? bhh_b : bhh_f;

  for (int idx = threadIdx.x; idx < Bm * Hm; idx += blockDim.x) {
    float v = state0[dir * Bm * Hm + idx];
    hf[idx] = v;
    hb0[idx] = f2bf(v);
  }
  __syncthreads();

  int lane = threadIdx.x & 31, wave = threadIdx.x >> 5;
  int half = lane >> 4, l15 = lane & 15;

  for (int s = 0; s < Tm; ++s) {
    int tt = dir ? (Tm - 1 - s) : s;
    __bf16* hcur  = (s & 1) ? hb1 : hb0;
    __bf16* hnext = (s & 1) ? hb0 : hb1;

    for (int u = 0; u < 4; ++u) {
      int unit = wave * 4 + u;          // 0..127
      int mt = unit >> 4;               // 0..7   (batch tile)
      int jt = unit & 15;               // 0..15  (column tile within H)
      int j16 = jt * 16 + l15;

      const __bf16* ap  = hcur + (mt * 16 + l15) * Hm;
      const __bf16* bpr = whh + (size_t)(0 * Hm + j16) * Hm;
      const __bf16* bpz = whh + (size_t)(1 * Hm + j16) * Hm;
      const __bf16* bpn = whh + (size_t)(2 * Hm + j16) * Hm;

      v8f accr = {}, accz = {}, accn = {};
#pragma unroll
      for (int kc = 0; kc < 8; ++kc) {
        int ka = kc * 32 + (half ? 8 : 0);
        v16bf a = make_frag(*(const v8bf*)(ap + ka), *(const v8bf*)(ap + ka + 16));
        int kb = kc * 32 + (half ? 8 : 0);
        v16bf br = make_frag(*(const v8bf*)(bpr + kb), *(const v8bf*)(bpr + kb + 16));
        accr = wmma_bf16(a, br, accr);
        v16bf bz = make_frag(*(const v8bf*)(bpz + kb), *(const v8bf*)(bpz + kb + 16));
        accz = wmma_bf16(a, bz, accz);
        v16bf bn = make_frag(*(const v8bf*)(bpn + kb), *(const v8bf*)(bpn + kb + 16));
        accn = wmma_bf16(a, bn, accn);
      }

      int j = j16;
      float br_ = bhh[j], bz_ = bhh[Hm + j], bn_ = bhh[2 * Hm + j];
#pragma unroll
      for (int i = 0; i < 8; ++i) {
        int b = mt * 16 + half * 8 + i;                     // batch row
        size_t it = ((size_t)tt * Bm + b) * 768;
        float gr = gi[it + j], gz = gi[it + Hm + j], gn = gi[it + 2 * Hm + j];
        float r = 1.f / (1.f + __expf(-(gr + accr[i] + br_)));
        float z = 1.f / (1.f + __expf(-(gz + accz[i] + bz_)));
        float n = tanhf(gn + r * (accn[i] + bn_));
        float hold = hf[b * Hm + j];
        float hnew = (1.f - z) * n + z * hold;
        hf[b * Hm + j]    = hnew;
        hnext[b * Hm + j] = f2bf(hnew);
        size_t o = ((size_t)b * Tm + tt) * 512 + dir * Hm + j;
        ow[o]   = hnew;
        owbf[o] = f2bf(hnew);
      }
    }
    __syncthreads();   // hnext fully written before next step's A-frag reads
  }

#if __has_builtin(__builtin_amdgcn_s_cluster_barrier)
  __builtin_amdgcn_s_cluster_barrier();   // NOP when not dispatched as cluster
#endif

  for (int idx = threadIdx.x; idx < Bm * Hm; idx += blockDim.x)
    state_out[dir * Bm * Hm + idx] = hf[idx];
}

// ---------------------------------------------------------------------------
// 5) attn logits: fused  sum_j tanh(ow@W + b)[i,j] * proj[j]
// ---------------------------------------------------------------------------
__global__ void __launch_bounds__(256)
k_zero_logits(float* __restrict__ logits) {
  int idx = blockIdx.x * 256 + threadIdx.x;
  if (idx < Bm * Tm) logits[idx] = 0.f;
}

__global__ void __launch_bounds__(256)
k_attn_logits(const __bf16* __restrict__ owbf, const __bf16* __restrict__ Wt,
              const float* __restrict__ b_word, const float* __restrict__ proj,
              float* __restrict__ logits) {
  const int NT = 32;
  int gwave = blockIdx.x * (blockDim.x >> 5) + (threadIdx.x >> 5);
  int mt = gwave / NT, nt = gwave - mt * NT;
  int lane = threadIdx.x & 31, half = lane >> 4, l15 = lane & 15;

  const __bf16* ap = owbf + (size_t)(mt * 16 + l15) * 512;
  int n = nt * 16 + l15;
  const __bf16* bp = Wt + (size_t)n * 512;
  __builtin_prefetch(ap, 0, 0);

  v8f acc = {};
#pragma unroll
  for (int kc = 0; kc < 16; ++kc) {
    int ka = kc * 32 + (half ? 8 : 0);
    v16bf a = make_frag(*(const v8bf*)(ap + ka), *(const v8bf*)(ap + ka + 16));
    int kb = kc * 32 + (half ? 8 : 0);
    v16bf b = make_frag(*(const v8bf*)(bp + kb), *(const v8bf*)(bp + kb + 16));
    acc = wmma_bf16(a, b, acc);
  }
  float bw = b_word[n], pj = proj[n];
  float part[8];
#pragma unroll
  for (int i = 0; i < 8; ++i) part[i] = tanhf(acc[i] + bw) * pj;
#pragma unroll
  for (int off = 1; off < 16; off <<= 1)
#pragma unroll
    for (int i = 0; i < 8; ++i) part[i] += __shfl_xor(part[i], off, 32);
  if (l15 == 0) {
    int rowbase = mt * 16 + half * 8;
#pragma unroll
    for (int i = 0; i < 8; ++i) atomicAdd(&logits[rowbase + i], part[i]);
  }
}

// ---------------------------------------------------------------------------
// 6) softmax over T per batch row
// ---------------------------------------------------------------------------
__global__ void __launch_bounds__(256)
k_softmax(const float* __restrict__ logits, float* __restrict__ out) {
  __shared__ float red[32];
  __shared__ float bcast;
  int b = blockIdx.x;
  const float* x = logits + b * Tm;

  float mx = -1e30f;
  for (int t = threadIdx.x; t < Tm; t += blockDim.x) mx = fmaxf(mx, x[t]);
  for (int off = 16; off; off >>= 1) mx = fmaxf(mx, __shfl_xor(mx, off, 32));
  if ((threadIdx.x & 31) == 0) red[threadIdx.x >> 5] = mx;
  __syncthreads();
  if (threadIdx.x == 0) {
    float m = -1e30f;
    for (int w = 0; w < (int)(blockDim.x >> 5); ++w) m = fmaxf(m, red[w]);
    bcast = m;
  }
  __syncthreads();
  mx = bcast;

  float s = 0.f;
  for (int t = threadIdx.x; t < Tm; t += blockDim.x) s += __expf(x[t] - mx);
  for (int off = 16; off; off >>= 1) s += __shfl_xor(s, off, 32);
  __syncthreads();
  if ((threadIdx.x & 31) == 0) red[threadIdx.x >> 5] = s;
  __syncthreads();
  if (threadIdx.x == 0) {
    float t2 = 0.f;
    for (int w = 0; w < (int)(blockDim.x >> 5); ++w) t2 += red[w];
    bcast = t2;
  }
  __syncthreads();
  float inv = 1.f / bcast;
  for (int t = threadIdx.x; t < Tm; t += blockDim.x)
    out[b * Tm + t] = __expf(x[t] - mx) * inv;
}

// ---------------------------------------------------------------------------
// 7) sent_vectors[b, j] = sum_t attn[b,t] * ow[b,t,j]
// ---------------------------------------------------------------------------
__global__ void __launch_bounds__(512)
k_sent(const float* __restrict__ attn, const float* __restrict__ ow,
       float* __restrict__ sent) {
  int b = blockIdx.x, j = threadIdx.x;     // j < 512
  float s = 0.f;
  for (int t = 0; t < Tm; ++t)
    s += attn[b * Tm + t] * ow[((size_t)b * Tm + t) * 512 + j];
  sent[b * 512 + j] = s;
}

// ---------------------------------------------------------------------------
extern "C" void kernel_launch(void* const* d_in, const int* in_sizes, int n_in,
                              void* d_out, int out_size, void* d_ws, size_t ws_size,
                              hipStream_t stream) {
  const int*   ids    = (const int*)  d_in[0];
  const float* state0 = (const float*)d_in[1];
  const float* emb    = (const float*)d_in[2];
  const float* wihf   = (const float*)d_in[3];
  const float* whhf   = (const float*)d_in[4];
  const float* bihf   = (const float*)d_in[5];
  const float* bhhf   = (const float*)d_in[6];
  const float* wihb   = (const float*)d_in[7];
  const float* whhb   = (const float*)d_in[8];
  const float* bihb   = (const float*)d_in[9];
  const float* bhhb   = (const float*)d_in[10];
  const float* Wword  = (const float*)d_in[11];
  const float* bword  = (const float*)d_in[12];
  const float* proj   = (const float*)d_in[13];

  float* out = (float*)d_out;
  float* o_sent  = out;                    // [128,512]
  float* o_state = out + 128 * 512;        // [2,128,256]
  float* o_attn  = o_state + 2 * 128 * 256;// [128,200]

  unsigned char* ws = (unsigned char*)d_ws;
  __bf16* xbf     = (__bf16*)(ws + OFF_X);
  __bf16* bwihf   = (__bf16*)(ws + OFF_WIHF);
  __bf16* bwhhf   = (__bf16*)(ws + OFF_WHHF);
  __bf16* bwihb   = (__bf16*)(ws + OFF_WIHB);
  __bf16* bwhhb   = (__bf16*)(ws + OFF_WHHB);
  __bf16* Wt      = (__bf16*)(ws + OFF_WT);
  float*  gi_f    = (float*) (ws + OFF_GIF);
  float*  gi_b    = (float*) (ws + OFF_GIB);
  float*  ow      = (float*) (ws + OFF_OW);
  __bf16* owbf    = (__bf16*)(ws + OFF_OWBF);
  float*  logits  = (float*) (ws + OFF_LOG);

  // 1) gather + convert x
  k_gather_x<<<Tm * Bm, 256, 0, stream>>>(ids, emb, xbf);
  // 2) convert weights
  k_conv_w<<<4096, 256, 0, stream>>>(wihf, whhf, wihb, whhb, Wword,
                                     bwihf, bwhhf, bwihb, bwhhb, Wt);
  // 3) gi GEMM (both dirs): 2*400*6 macro-tile blocks, TDM+async staged
  k_gemm_gi<<<4800, 256, 0, stream>>>(xbf, bwihf, bwihb, bihf, bihb, gi_f, gi_b);
  // zero attn logits before scan produces ow
  k_zero_logits<<<100, 256, 0, stream>>>(logits);
  // 4) GRU scan, one persistent workgroup per direction, 256KB dynamic LDS
  k_gru_scan<<<2, 1024, 262144, stream>>>(gi_f, gi_b, bwhhf, bwhhb, bhhf, bhhb,
                                          state0, ow, owbf, o_state);
  // 5) attention logits: 1600*32 waves, 8 waves/block
  k_attn_logits<<<6400, 256, 0, stream>>>(owbf, Wt, bword, proj, logits);
  // 6) softmax over T
  k_softmax<<<Bm, 256, 0, stream>>>(logits, o_attn);
  // 7) weighted sum
  k_sent<<<Bm, 512, 0, stream>>>(o_attn, ow, o_sent);
}
